// Parse_Graph_Module_4020089389664
// MI455X (gfx1250) — matrix operand order repaired and checked by reading, not compiled, not run on hardware
//
#include <hip/hip_runtime.h>
#include <stdint.h>

// ---------- types ----------
typedef __attribute__((ext_vector_type(16))) __bf16 bf16x16;
typedef __attribute__((ext_vector_type(2)))  __bf16 bf16x2;
typedef __attribute__((ext_vector_type(8)))  float  floatx8;

union FragAB { bf16x16 v; uint4 q[2]; };

static constexpr int Bn = 8, Nn = 2048, Hn = 256;
static constexpr int LDW = 40;                 // padded LDS row stride (bf16 units)

// ---------- fp32 -> bf16 (RNE) ----------
#if __has_builtin(__builtin_amdgcn_cvt_pk_bf16_f32)
__device__ __forceinline__ uint32_t pkbf(float lo, float hi) {
    bf16x2 r = __builtin_amdgcn_cvt_pk_bf16_f32(lo, hi);
    return __builtin_bit_cast(uint32_t, r);
}
__device__ __forceinline__ uint16_t f2bf(float f) {
    return (uint16_t)pkbf(f, 0.f);
}
#else
__device__ __forceinline__ uint32_t pkbf(float lo, float hi) {
    bf16x2 r;
    r.x = (__bf16)lo;
    r.y = (__bf16)hi;
    return __builtin_bit_cast(uint32_t, r);
}
__device__ __forceinline__ uint16_t f2bf(float f) {
    __bf16 h = (__bf16)f;
    return __builtin_bit_cast(uint16_t, h);
}
#endif

// ---------- CDNA5 async global->LDS copy (ASYNCcnt-tracked, no data VGPRs) ----------
template <int OFF>
__device__ __forceinline__ void async_b128(uint32_t lds, uint64_t gaddr) {
    asm volatile("global_load_async_to_lds_b128 %0, %1, off offset:%c2"
                 :: "v"(lds), "v"(gaddr), "i"(OFF) : "memory");
}
__device__ __forceinline__ void wait_async0() {
#if __has_builtin(__builtin_amdgcn_s_wait_asynccnt)
    __builtin_amdgcn_s_wait_asynccnt(0);
#else
    asm volatile("s_wait_asynccnt 0x0" ::: "memory");
#endif
}
__device__ __forceinline__ uint32_t lds_addr(const void* p) {
    return (uint32_t)(uintptr_t)p;             // low 32 bits = LDS offset (flat aperture rule)
}

// ---------- WMMA fragment loaders (ISA 7.12.2 layouts, padded stride) ----------
__device__ __forceinline__ FragAB load_a_frag(const uint16_t* At, int row, int kh) {
    FragAB a;
    const uint16_t* base = At + row * LDW;
    a.q[0] = *(const uint4*)(base + 8 * kh);
    a.q[1] = *(const uint4*)(base + 16 + 8 * kh);
    return a;
}
__device__ __forceinline__ FragAB load_b_frag(const uint16_t* Bt, int n, int kh) {
    FragAB b;
    const uint16_t* base = Bt + n * LDW;
    b.q[0] = *(const uint4*)(base + 8 * kh);
    b.q[1] = *(const uint4*)(base + 16 + 8 * kh);
    return b;
}

// ---------- 1) dinv = rsqrt(row-sum of graph) ----------
__global__ void rowsum_dinv_kernel(const float* __restrict__ graph, float* __restrict__ dinv) {
    const int row = blockIdx.x;
    const float* src = graph + (size_t)row * Nn;
    float s = 0.f;
#pragma unroll
    for (int p = 0; p < 2; ++p) {
        float4 f = *(const float4*)(src + (threadIdx.x + p * 256) * 4);
        s += f.x + f.y + f.z + f.w;
    }
    __shared__ float red[256];
    red[threadIdx.x] = s;
    __syncthreads();
    for (int off = 128; off > 0; off >>= 1) {
        if (threadIdx.x < off) red[threadIdx.x] += red[threadIdx.x + off];
        __syncthreads();
    }
    if (threadIdx.x == 0) dinv[row] = rsqrtf(red[0]);
}

// ---------- 2) fp32 -> bf16 (weights) ----------
__global__ void cvt_bf16_kernel(const float* __restrict__ src, uint16_t* __restrict__ dst, int n) {
    int i = (blockIdx.x * 256 + threadIdx.x) * 2;
    if (i < n) *(uint32_t*)(dst + i) = pkbf(src[i], src[i + 1]);
}

// ---------- 3) node -> node_bf16 (normal) + dinv*node (panel-transposed) ----------
__global__ void __launch_bounds__(256)
prep_node_kernel(const float* __restrict__ node, const float* __restrict__ dinv,
                 uint16_t* __restrict__ node_b, uint16_t* __restrict__ nscT) {
    const int b = blockIdx.z, p = blockIdx.x;
    const int tid = threadIdx.x;
    __shared__ __align__(16) uint16_t Lt[256 * 32];    // [h][m_local]

    const int ml  = tid >> 3;
    const int hch = (tid & 7) * 32;
    const int m   = p * 32 + ml;
    const float* src = node + ((size_t)b * Nn + m) * Hn + hch;
    const float s = dinv[(size_t)b * Nn + m];

    uint16_t* nb = node_b + ((size_t)b * Nn + m) * Hn + hch;
#pragma unroll
    for (int q = 0; q < 8; ++q) {
        float4 f = *(const float4*)(src + q * 4);
        *(uint2*)(nb + q * 4) = (uint2){pkbf(f.x, f.y), pkbf(f.z, f.w)};
        Lt[(hch + q * 4 + 0) * 32 + ml] = f2bf(f.x * s);
        Lt[(hch + q * 4 + 1) * 32 + ml] = f2bf(f.y * s);
        Lt[(hch + q * 4 + 2) * 32 + ml] = f2bf(f.z * s);
        Lt[(hch + q * 4 + 3) * 32 + ml] = f2bf(f.w * s);
    }
    __syncthreads();
    uint16_t* dst = nscT + (((size_t)b * 64 + p) * 256 + tid) * 32;
#pragma unroll
    for (int q = 0; q < 4; ++q)
        *(uint4*)(dst + q * 8) = *(const uint4*)(&Lt[tid * 32 + q * 8]);
}

// ---------- 4) aggregation GEMM: out = bf16( dinv[row] * (graph @ B) ) ----------
__global__ void __launch_bounds__(256, 1)
agg_gemm_kernel(const float* __restrict__ graph, const uint16_t* __restrict__ BT,
                const float* __restrict__ dinv, uint16_t* __restrict__ out) {
    const int b   = blockIdx.z;
    const int m0  = blockIdx.x * 64;
    const int tid = threadIdx.x;
    const int wave = tid >> 5, lane = tid & 31;
    const int wr = wave >> 1, wc = wave & 1;
    const int ll = lane & 15, kh = lane >> 4;

    __shared__ __align__(16) uint16_t At[2][64 * LDW];
    __shared__ __align__(16) uint16_t Bt[2][256 * LDW];

    const float*    gA  = graph + ((size_t)b * Nn + m0) * Nn;
    const uint16_t* gBT = BT + (size_t)b * Nn * Hn;
    const float*    dv  = dinv + (size_t)b * Nn;

    const int arow = tid >> 3, aseg = tid & 7;

    const uint32_t ldsB[2] = {lds_addr(&Bt[0][tid * LDW]), lds_addr(&Bt[1][tid * LDW])};

    floatx8 acc[8];
    const floatx8 z = {0.f, 0.f, 0.f, 0.f, 0.f, 0.f, 0.f, 0.f};
#pragma unroll
    for (int c = 0; c < 8; ++c) acc[c] = z;

    float4 av[2];

    auto fill_b_async = [&](int it, int buf) {
        uint64_t g = (uint64_t)(uintptr_t)(gBT + ((size_t)it * 256 + tid) * 32);
        async_b128<0>(ldsB[buf], g);
        async_b128<16>(ldsB[buf], g);
        async_b128<32>(ldsB[buf], g);
        async_b128<48>(ldsB[buf], g);
    };
    auto load_a = [&](int it) {
        const int k0 = it * 32;
#pragma unroll
        for (int p = 0; p < 2; ++p)
            av[p] = *(const float4*)(gA + (size_t)(arow + p * 32) * Nn + k0 + aseg * 4);
    };
    auto store_a = [&](int buf) {
#pragma unroll
        for (int p = 0; p < 2; ++p) {
            uint16_t* dst = &At[buf][(arow + p * 32) * LDW + aseg * 4];
            *(uint2*)dst = (uint2){pkbf(av[p].x, av[p].y), pkbf(av[p].z, av[p].w)};
        }
    };

    const int NK = Nn / 32;
    fill_b_async(0, 0);
    load_a(0);
    store_a(0);
    for (int i = 0; i < NK; ++i) {
        wait_async0();
        __syncthreads();
        const int cur = i & 1;
        const bool more = (i + 1) < NK;
        if (more) { fill_b_async(i + 1, cur ^ 1); load_a(i + 1); }
        // bulk-preload all fragments (distinct regs -> pipelined ds_loads, no WAR nops)
        FragAB a = load_a_frag(At[cur], wr * 16 + ll, kh);
        FragAB bb[8];
#pragma unroll
        for (int c = 0; c < 8; ++c)
            bb[c] = load_b_frag(Bt[cur], wc * 128 + c * 16 + ll, kh);
#pragma unroll
        for (int c = 0; c < 8; ++c)
            acc[c] = __builtin_amdgcn_wmma_f32_16x16x32_bf16(
                false, a.v, false, bb[c].v, (short)0, acc[c], false, false);
        if (more) store_a(cur ^ 1);            // cvt + ds_store after WMMA issue
    }

#pragma unroll
    for (int c = 0; c < 8; ++c) {
        int n = wc * 128 + c * 16 + ll;
#pragma unroll
        for (int v = 0; v < 8; ++v) {
            int m = m0 + wr * 16 + kh * 8 + v;
            out[((size_t)b * Nn + m) * Hn + n] = f2bf(acc[c][v] * dv[m]);
        }
    }
}

// ---------- 5) FC GEMM: out = [relu](A @ W^T + bias) [* dinv] ----------
template <bool RELU, bool SCALE, bool TRANSOUT, typename OutT>
__global__ void __launch_bounds__(256, 1)
fc_gemm_kernel(const uint16_t* __restrict__ A1, const uint16_t* __restrict__ A2,
               int K1, int Ktot, const uint16_t* __restrict__ W,
               const float* __restrict__ bias, const float* __restrict__ dinv,
               OutT* __restrict__ out) {
    const int b   = blockIdx.z;
    const int m0  = blockIdx.x * 64;
    const int tid = threadIdx.x;
    const int wave = tid >> 5, lane = tid & 31;
    const int wr = wave >> 1, wc = wave & 1;
    const int ll = lane & 15, kh = lane >> 4;
    const int K2 = Ktot - K1;

    __shared__ __align__(16) uint16_t At[2][64 * LDW];
    __shared__ __align__(16) uint16_t Bt[2][256 * LDW];

    const int arow = tid >> 2, aseg = tid & 3;
    const uint32_t ldsA[2] = {lds_addr(&At[0][arow * LDW + aseg * 8]),
                              lds_addr(&At[1][arow * LDW + aseg * 8])};
    const uint32_t ldsB[2] = {lds_addr(&Bt[0][tid * LDW]), lds_addr(&Bt[1][tid * LDW])};

    floatx8 acc[8];
    const floatx8 z = {0.f, 0.f, 0.f, 0.f, 0.f, 0.f, 0.f, 0.f};
#pragma unroll
    for (int c = 0; c < 8; ++c) acc[c] = z;

    auto fill_async = [&](int it, int buf) {
        const int koff = it * 32;
        const uint16_t* Asrc; int lda, kloc;
        if (koff < K1) { Asrc = A1 + (size_t)b * Nn * K1; lda = K1; kloc = koff; }
        else           { Asrc = A2 + (size_t)b * Nn * K2; lda = K2; kloc = koff - K1; }
        uint64_t ga = (uint64_t)(uintptr_t)(Asrc + (size_t)(m0 + arow) * lda + kloc + aseg * 8);
        async_b128<0>(ldsA[buf], ga);
        uint64_t gw = (uint64_t)(uintptr_t)(W + (size_t)tid * Ktot + koff);
        async_b128<0>(ldsB[buf], gw);
        async_b128<16>(ldsB[buf], gw);
        async_b128<32>(ldsB[buf], gw);
        async_b128<48>(ldsB[buf], gw);
    };

    const int NK = Ktot / 32;
    fill_async(0, 0);
    for (int i = 0; i < NK; ++i) {
        wait_async0();
        __syncthreads();
        const int cur = i & 1;
        const bool more = (i + 1) < NK;
        if (more) fill_async(i + 1, cur ^ 1);
        FragAB a = load_a_frag(At[cur], wr * 16 + ll, kh);
        FragAB bb[8];
#pragma unroll
        for (int c = 0; c < 8; ++c)
            bb[c] = load_b_frag(Bt[cur], wc * 128 + c * 16 + ll, kh);
#pragma unroll
        for (int c = 0; c < 8; ++c)
            acc[c] = __builtin_amdgcn_wmma_f32_16x16x32_bf16(
                false, a.v, false, bb[c].v, (short)0, acc[c], false, false);
    }

    const int mbase = wr * 16 + kh * 8;
#pragma unroll
    for (int c = 0; c < 8; ++c) {
        int n = wc * 128 + c * 16 + ll;
        float bv = bias[n];
        float vals[8];
#pragma unroll
        for (int v = 0; v < 8; ++v) {
            int m = m0 + mbase + v;
            float val = acc[c][v] + bv;
            if (RELU) val = val > 0.f ? val : 0.f;
            if (SCALE) val *= dinv[(size_t)b * Nn + m];
            vals[v] = val;
        }
        if constexpr (TRANSOUT) {
            const int panel = (m0 + mbase) >> 5, kl0 = mbase & 31;
            uint4 q = {pkbf(vals[0], vals[1]), pkbf(vals[2], vals[3]),
                       pkbf(vals[4], vals[5]), pkbf(vals[6], vals[7])};
            *(uint4*)((uint16_t*)out + (((size_t)b * 64 + panel) * 256 + n) * 32 + kl0) = q;
        } else {
#pragma unroll
            for (int v = 0; v < 8; ++v) {
                int m = m0 + mbase + v;
                OutT* dst = out + ((size_t)b * Nn + m) * Hn + n;
                if constexpr (sizeof(OutT) == 2) *dst = (OutT)f2bf(vals[v]);
                else                             *dst = (OutT)vals[v];
            }
        }
    }
}

// ---------- launch ----------
extern "C" void kernel_launch(void* const* d_in, const int* in_sizes, int n_in,
                              void* d_out, int out_size, void* d_ws, size_t ws_size,
                              hipStream_t stream) {
    const float* node  = (const float*)d_in[0];
    const float* graph = (const float*)d_in[1];
    const float* W1    = (const float*)d_in[2];
    const float* b1    = (const float*)d_in[3];
    const float* W2    = (const float*)d_in[4];
    const float* b2    = (const float*)d_in[5];
    const float* Wout  = (const float*)d_in[6];
    const float* bout  = (const float*)d_in[7];
    float* outp = (float*)d_out;

    char* ws = (char*)d_ws;
    size_t off = 0;
    auto alloc = [&](size_t bytes) { size_t r = off; off += (bytes + 255) & ~(size_t)255; return r; };
    float*    dinvp  = (float*)(ws + alloc((size_t)Bn * Nn * 4));
    uint16_t* W1b    = (uint16_t*)(ws + alloc((size_t)Hn * Hn * 2));
    uint16_t* W2b    = (uint16_t*)(ws + alloc((size_t)Hn * Hn * 2));
    uint16_t* Woutb  = (uint16_t*)(ws + alloc((size_t)Hn * 2 * Hn * 2));
    uint16_t* node_b = (uint16_t*)(ws + alloc((size_t)Bn * Nn * Hn * 2));
    uint16_t* nscT   = (uint16_t*)(ws + alloc((size_t)Bn * Nn * Hn * 2));
    uint16_t* tbuf   = (uint16_t*)(ws + alloc((size_t)Bn * Nn * Hn * 2));
    uint16_t* hT     = (uint16_t*)(ws + alloc((size_t)Bn * Nn * Hn * 2));
    uint16_t* hbuf   = (uint16_t*)(ws + alloc((size_t)Bn * Nn * Hn * 2));

    const dim3 blk(256);
    const dim3 ggrid(Nn / 64, 1, Bn);

    rowsum_dinv_kernel<<<Bn * Nn, blk, 0, stream>>>(graph, dinvp);
    cvt_bf16_kernel<<<(Hn * Hn / 2) / 256, blk, 0, stream>>>(W1, W1b, Hn * Hn);
    cvt_bf16_kernel<<<(Hn * Hn / 2) / 256, blk, 0, stream>>>(W2, W2b, Hn * Hn);
    cvt_bf16_kernel<<<(Hn * Hn) / 256, blk, 0, stream>>>(Wout, Woutb, Hn * 2 * Hn);
    prep_node_kernel<<<dim3(Nn / 32, 1, Bn), blk, 0, stream>>>(node, dinvp, node_b, nscT);
    // t1 = dinv ⊙ (graph @ (dinv⊙node))
    agg_gemm_kernel<<<ggrid, blk, 0, stream>>>(graph, nscT, dinvp, tbuf);
    // h1s = dinv ⊙ relu(t1 @ W1^T + b1)   [panel-transposed]
    fc_gemm_kernel<true, true, true, uint16_t><<<ggrid, blk, 0, stream>>>(
        tbuf, tbuf, Hn, Hn, W1b, b1, dinvp, hT);
    // t2 = dinv ⊙ (graph @ h1s)
    agg_gemm_kernel<<<ggrid, blk, 0, stream>>>(graph, hT, dinvp, tbuf);
    // h2 = relu(t2 @ W2^T + b2)
    fc_gemm_kernel<true, false, false, uint16_t><<<ggrid, blk, 0, stream>>>(
        tbuf, tbuf, Hn, Hn, W2b, b2, dinvp, hbuf);
    // out = relu([node, h2] @ Wout^T + bout)
    fc_gemm_kernel<true, false, false, float><<<ggrid, blk, 0, stream>>>(
        node_b, hbuf, Hn, 2 * Hn, Woutb, bout, dinvp, outp);
}
